// BlendshapeTransformer_86363202388177
// MI455X (gfx1250) — compile-verified, matrix operand order checked by reading, not run on hardware
//
#include <hip/hip_runtime.h>
#include <hip/hip_bf16.h>

typedef __attribute__((ext_vector_type(16))) _Float16 v16h;
typedef __attribute__((ext_vector_type(8)))  _Float16 v8h;
typedef __attribute__((ext_vector_type(8)))  float    v8f;

#define TOKS   8192     // B*T
#define TSEQ   2048
#define DMODEL 256
#define NHEAD  8
#define DHEAD  32
#define FFN    512
#define WIN    43       // PAST(30) + FUT(12) + 1

// ---------------------------------------------------------------------------
// Weight convert: W f32 [K x N] -> Wt f16 [Np x Kp] (transposed, zero padded).
// N-major layout so a WMMA B-fragment lane (output column n) reads contiguous K.
// ---------------------------------------------------------------------------
__global__ void wconvT_kernel(const float* __restrict__ W, _Float16* __restrict__ Wt,
                              int K, int N, int Kp, int Np) {
  int i = blockIdx.x * blockDim.x + threadIdx.x;
  int total = Kp * Np;
  if (i >= total) return;
  int n = i / Kp, k = i - n * Kp;
  float v = (k < K && n < N) ? W[(size_t)k * N + n] : 0.f;
  Wt[i] = (_Float16)v;
}

__global__ void f32_to_f16_kernel(const float* __restrict__ X, _Float16* __restrict__ Y, int n) {
  int i = blockIdx.x * blockDim.x + threadIdx.x;
  if (i < n) Y[i] = (_Float16)X[i];
}

// ---------------------------------------------------------------------------
// Build encoder input: concat(af[120], pt[1], ph_emb[pi][32], sp_emb[si][8])
// zero-padded to 192 columns, f16.
// ---------------------------------------------------------------------------
__global__ void build_x0_kernel(const float* __restrict__ af, const float* __restrict__ pt,
                                const int* __restrict__ pi, const int* __restrict__ si,
                                const float* __restrict__ ph_emb, const float* __restrict__ sp_emb,
                                _Float16* __restrict__ X0) {
  int i = blockIdx.x * blockDim.x + threadIdx.x;
  if (i >= TOKS * 192) return;
  int m = i / 192, c = i - m * 192;
  float v = 0.f;
  if (c < 120)       v = af[(size_t)m * 120 + c];
  else if (c == 120) v = pt[m];
  else if (c < 153)  v = ph_emb[pi[m] * 32 + (c - 121)];
  else if (c < 161)  v = sp_emb[si[m] * 8 + (c - 153)];
  X0[i] = (_Float16)v;
}

// ---------------------------------------------------------------------------
// Fragment load: 16 halves per lane, per the CDNA5 16-bit A/B WMMA layout
// (elements 0..7 at k0, elements 8..15 at k0+16).
// ---------------------------------------------------------------------------
__device__ __forceinline__ v16h ldfrag(const _Float16* p) {
  v8h lo = *(const v8h*)p;
  v8h hi = *(const v8h*)(p + 16);
  v16h r;
#pragma unroll
  for (int i = 0; i < 8; ++i) { r[i] = lo[i]; r[i + 8] = hi[i]; }
  return r;
}

__device__ __forceinline__ v8f wmma16(v16h a, v16h b, v8f c) {
  return __builtin_amdgcn_wmma_f32_16x16x32_f16(false, a, false, b, (short)0, c, false, false);
}

// ---------------------------------------------------------------------------
// Generic WMMA GEMM with 2x4 register tiling: each wave computes a 32x64
// macro-tile (8 x v_wmma_f32_16x16x32_f16 per 32-wide k-step; 12 b128 loads).
// KP is compile-time so all row strides fold into 24-bit instruction offsets
// (2 address pointers total) and register pressure stays below spill level.
//   C[M x N] = act(A[M x KP] @ Wt^T + bias) (+ residual)
//   A  : f16, row-major, stride KP (KP % 32 == 0); M % 32 == 0
//   Wt : f16, [Np x KP], row n = output column n of W; Np % 64 == 0
// act: 0=none 1=relu 2=gelu(exact) 3=sigmoid
// ---------------------------------------------------------------------------
template<int KP>
__global__ void gemm_wmma_kernel(const _Float16* __restrict__ A, const _Float16* __restrict__ Wt,
                                 const float* __restrict__ bias,
                                 const float* residual, float* Cf, _Float16* Ch,
                                 int M, int N, int Np, int act) {
  int wave = (blockIdx.x * blockDim.x + threadIdx.x) >> 5;
  int lane = threadIdx.x & 31;
  int tilesN = Np >> 6;                      // 64-wide macro-tiles
  int tiles  = (M >> 5) * tilesN;            // 32-tall macro-tiles
  if (wave >= tiles) return;                 // wave-uniform: EXEC stays all-1s

  int mt = (wave / tilesN) << 5;
  int nt = (wave % tilesN) << 6;
  int khi = (lane >> 4) << 3;                // 0 for lanes 0-15, 8 for lanes 16-31

  const _Float16* Ap = A  + (size_t)(mt + (lane & 15)) * KP + khi;
  const _Float16* Bp = Wt + (size_t)(nt + (lane & 15)) * KP + khi;

  v8f acc[8];
#pragma unroll
  for (int i = 0; i < 8; ++i) acc[i] = (v8f){};

#pragma unroll 2
  for (int ks = 0; ks < KP; ks += 32) {
    v16h a0 = ldfrag(Ap + ks);
    v16h a1 = ldfrag(Ap + ks + 16 * KP);     // constant offsets from 2 pointers
    v16h b0 = ldfrag(Bp + ks);
    v16h b1 = ldfrag(Bp + ks + 16 * KP);
    v16h b2 = ldfrag(Bp + ks + 32 * KP);
    v16h b3 = ldfrag(Bp + ks + 48 * KP);
    acc[0] = wmma16(a0, b0, acc[0]);
    acc[1] = wmma16(a0, b1, acc[1]);
    acc[2] = wmma16(a0, b2, acc[2]);
    acc[3] = wmma16(a0, b3, acc[3]);
    acc[4] = wmma16(a1, b0, acc[4]);
    acc[5] = wmma16(a1, b1, acc[5]);
    acc[6] = wmma16(a1, b2, acc[6]);
    acc[7] = wmma16(a1, b3, acc[7]);
  }

  int mrow = ((lane >> 4) << 3);             // D layout: M = g + (lane>=16 ? 8 : 0)
  int ncol = (lane & 15);
#pragma unroll
  for (int mi = 0; mi < 2; ++mi) {
#pragma unroll
    for (int ni = 0; ni < 4; ++ni) {
      int n = nt + ni * 16 + ncol;
      if (n >= N) continue;
      float bv = bias ? bias[n] : 0.f;
      int mbase = mt + mi * 16 + mrow;
      v8f a = acc[mi * 4 + ni];
#pragma unroll
      for (int g = 0; g < 8; ++g) {
        int m = mbase + g;
        float v = a[g] + bv;
        if (act == 1)      v = fmaxf(v, 0.f);
        else if (act == 2) v = 0.5f * v * (1.f + erff(v * 0.70710678118654752f));
        else if (act == 3) v = 1.f / (1.f + __expf(-v));
        if (residual) v += residual[(size_t)m * N + n];
        if (Cf) Cf[(size_t)m * N + n] = v;
        if (Ch) Ch[(size_t)m * N + n] = (_Float16)v;
      }
    }
  }
}

// ---------------------------------------------------------------------------
// LayerNorm (wave per token, D=256, 8 values per lane), f32 in -> f16 out.
// ---------------------------------------------------------------------------
__global__ void ln_f16_kernel(const float* __restrict__ X, const float* __restrict__ g,
                              const float* __restrict__ b, _Float16* __restrict__ H) {
  int wave = (blockIdx.x * blockDim.x + threadIdx.x) >> 5;
  int lane = threadIdx.x & 31;
  if (wave >= TOKS) return;
  const float* row = X + (size_t)wave * DMODEL;
  float vals[8], s = 0.f, sq = 0.f;
#pragma unroll
  for (int i = 0; i < 8; ++i) { int c = lane + 32 * i; float v = row[c]; vals[i] = v; s += v; sq += v * v; }
#pragma unroll
  for (int off = 16; off; off >>= 1) { s += __shfl_xor(s, off, 32); sq += __shfl_xor(sq, off, 32); }
  float mean = s * (1.f / DMODEL);
  float var  = sq * (1.f / DMODEL) - mean * mean;
  float rstd = rsqrtf(var + 1e-5f);
  _Float16* out = H + (size_t)wave * DMODEL;
#pragma unroll
  for (int i = 0; i < 8; ++i) {
    int c = lane + 32 * i;
    out[c] = (_Float16)((vals[i] - mean) * rstd * g[c] + b[c]);
  }
}

// ---------------------------------------------------------------------------
// Encoder epilogue: LN -> exact GELU -> + sinusoidal pos-enc, f32 out.
// ---------------------------------------------------------------------------
__global__ void enc_post_kernel(const float* __restrict__ pre, const float* __restrict__ g,
                                const float* __restrict__ b, float* __restrict__ x) {
  int wave = (blockIdx.x * blockDim.x + threadIdx.x) >> 5;
  int lane = threadIdx.x & 31;
  if (wave >= TOKS) return;
  int t = wave & (TSEQ - 1);
  const float* row = pre + (size_t)wave * DMODEL;
  float vals[8], s = 0.f, sq = 0.f;
#pragma unroll
  for (int i = 0; i < 8; ++i) { int c = lane + 32 * i; float v = row[c]; vals[i] = v; s += v; sq += v * v; }
#pragma unroll
  for (int off = 16; off; off >>= 1) { s += __shfl_xor(s, off, 32); sq += __shfl_xor(sq, off, 32); }
  float mean = s * (1.f / DMODEL);
  float var  = sq * (1.f / DMODEL) - mean * mean;
  float rstd = rsqrtf(var + 1e-5f);
  float* out = x + (size_t)wave * DMODEL;
#pragma unroll
  for (int i = 0; i < 8; ++i) {
    int c = lane + 32 * i;
    float v = (vals[i] - mean) * rstd * g[c] + b[c];
    v = 0.5f * v * (1.f + erff(v * 0.70710678118654752f));        // exact gelu
    int ce   = c & ~1;                                            // arange(0,D,2) value
    float fr = __expf((float)ce * (-9.210340371976184f / 256.f)); // exp(-ln(1e4)*ce/D)
    float ang = (float)t * fr;
    v += (c & 1) ? __cosf(ang) : __sinf(ang);
    out[c] = v;
  }
}

// ---------------------------------------------------------------------------
// Banded attention: one thread per (token, head). Window [-30, +12], DH=32.
// Q/K/V f16 [TOKS x 256] (head h = cols h*32..h*32+31) -> O f16 same layout.
// ---------------------------------------------------------------------------
__global__ void banded_attn_kernel(const _Float16* __restrict__ Q, const _Float16* __restrict__ K,
                                   const _Float16* __restrict__ V, _Float16* __restrict__ O) {
  int idx = blockIdx.x * blockDim.x + threadIdx.x;
  if (idx >= TOKS * NHEAD) return;
  int h = idx & (NHEAD - 1);
  int m = idx >> 3;                       // b*TSEQ + t
  int t = m & (TSEQ - 1);
  int bbase = m - t;

  const _Float16* q = Q + (size_t)m * DMODEL + h * DHEAD;
  float qf[DHEAD];
#pragma unroll
  for (int d = 0; d < DHEAD; ++d) qf[d] = (float)q[d] * 0.17677669529663687f;  // 1/sqrt(32)

  float s[WIN];
  float mx = -1e30f;
#pragma unroll
  for (int jj = 0; jj < WIN; ++jj) {
    int j = t - 30 + jj;
    float dot = -1e30f;
    if (j >= 0 && j < TSEQ) {
      const _Float16* kr = K + (size_t)(bbase + j) * DMODEL + h * DHEAD;
      dot = 0.f;
#pragma unroll
      for (int d = 0; d < DHEAD; ++d) dot += qf[d] * (float)kr[d];
    }
    s[jj] = dot;
    mx = fmaxf(mx, dot);
  }

  float denom = 0.f;
#pragma unroll
  for (int jj = 0; jj < WIN; ++jj) {
    int j = t - 30 + jj;
    float e = (j >= 0 && j < TSEQ) ? __expf(s[jj] - mx) : 0.f;
    s[jj] = e;
    denom += e;
  }

  float acc[DHEAD];
#pragma unroll
  for (int d = 0; d < DHEAD; ++d) acc[d] = 0.f;
#pragma unroll
  for (int jj = 0; jj < WIN; ++jj) {
    int j = t - 30 + jj;
    if (j < 0 || j >= TSEQ) continue;
    float w = s[jj];
    const _Float16* vr = V + (size_t)(bbase + j) * DMODEL + h * DHEAD;
#pragma unroll
    for (int d = 0; d < DHEAD; ++d) acc[d] += w * (float)vr[d];
  }

  float inv = 1.f / denom;
  _Float16* out = O + (size_t)m * DMODEL + h * DHEAD;
#pragma unroll
  for (int d = 0; d < DHEAD; ++d) out[d] = (_Float16)(acc[d] * inv);
}

// ---------------------------------------------------------------------------
// Launcher
// ---------------------------------------------------------------------------
extern "C" void kernel_launch(void* const* d_in, const int* in_sizes, int n_in,
                              void* d_out, int out_size, void* d_ws, size_t ws_size,
                              hipStream_t stream) {
  (void)in_sizes; (void)n_in; (void)out_size; (void)ws_size;
  const int TB = 256;

  // ---- inputs (setup_inputs dict order, nested dicts flattened in order) ----
  const float* af      = (const float*)d_in[0];
  const int*   pi      = (const int*)  d_in[1];
  const float* pt      = (const float*)d_in[2];
  const int*   si      = (const int*)  d_in[3];
  const float* ph_emb  = (const float*)d_in[4];
  const float* sp_emb  = (const float*)d_in[5];
  const float* enc_w   = (const float*)d_in[6];
  const float* enc_b   = (const float*)d_in[7];
  const float* encln_s = (const float*)d_in[8];
  const float* encln_b = (const float*)d_in[9];
  // layers: base 10 + 16*l : wq bq wk bk wv bv wo bo ln1s ln1b ln2s ln2b w1 b1 w2 b2
  const float* hw1 = (const float*)d_in[74];
  const float* hb1 = (const float*)d_in[75];
  const float* hw2 = (const float*)d_in[76];
  const float* hb2 = (const float*)d_in[77];

  // ---- workspace layout ----
  char* wsp = (char*)d_ws;
  size_t off = 0;
  auto alloc = [&](size_t bytes) -> void* {
    off = (off + 255) & ~(size_t)255;
    void* p = wsp + off;
    off += bytes;
    return p;
  };
  _Float16* X0  = (_Float16*)alloc((size_t)TOKS * 192 * 2);
  float*    x   = (float*)   alloc((size_t)TOKS * DMODEL * 4);
  float*    pre = (float*)   alloc((size_t)TOKS * DMODEL * 4);
  _Float16* H   = (_Float16*)alloc((size_t)TOKS * DMODEL * 2);
  _Float16* Qb  = (_Float16*)alloc((size_t)TOKS * DMODEL * 2);
  _Float16* Kb  = (_Float16*)alloc((size_t)TOKS * DMODEL * 2);
  _Float16* Vb  = (_Float16*)alloc((size_t)TOKS * DMODEL * 2);
  _Float16* Ob  = (_Float16*)alloc((size_t)TOKS * DMODEL * 2);
  _Float16* Fb  = (_Float16*)alloc((size_t)TOKS * FFN * 2);
  _Float16* Y1  = (_Float16*)alloc((size_t)TOKS * 128 * 2);
  _Float16* encT = (_Float16*)alloc((size_t)256 * 192 * 2);
  _Float16 *wqT[4], *wkT[4], *wvT[4], *woT[4], *w1T[4], *w2T[4];
  for (int l = 0; l < 4; ++l) {
    wqT[l] = (_Float16*)alloc((size_t)256 * 256 * 2);
    wkT[l] = (_Float16*)alloc((size_t)256 * 256 * 2);
    wvT[l] = (_Float16*)alloc((size_t)256 * 256 * 2);
    woT[l] = (_Float16*)alloc((size_t)256 * 256 * 2);
    w1T[l] = (_Float16*)alloc((size_t)512 * 256 * 2);
    w2T[l] = (_Float16*)alloc((size_t)256 * 512 * 2);
  }
  _Float16* h1T = (_Float16*)alloc((size_t)128 * 256 * 2);
  _Float16* h2T = (_Float16*)alloc((size_t)64 * 128 * 2);

  auto wconv = [&](const float* W, _Float16* Wt, int K, int N, int Kp, int Np) {
    int total = Kp * Np;
    wconvT_kernel<<<(total + TB - 1) / TB, TB, 0, stream>>>(W, Wt, K, N, Kp, Np);
  };
  auto gemm = [&](const _Float16* A, const _Float16* Wt, const float* bias,
                  const float* res, float* Cf, _Float16* Ch,
                  int M, int N, int Kp, int Np, int act) {
    int tiles = (M >> 5) * (Np >> 6);        // 32x64 macro-tiles, one wave each
    int blocks = (tiles * 32 + TB - 1) / TB;
    switch (Kp) {
      case 128: gemm_wmma_kernel<128><<<blocks, TB, 0, stream>>>(A, Wt, bias, res, Cf, Ch, M, N, Np, act); break;
      case 192: gemm_wmma_kernel<192><<<blocks, TB, 0, stream>>>(A, Wt, bias, res, Cf, Ch, M, N, Np, act); break;
      case 256: gemm_wmma_kernel<256><<<blocks, TB, 0, stream>>>(A, Wt, bias, res, Cf, Ch, M, N, Np, act); break;
      case 512: gemm_wmma_kernel<512><<<blocks, TB, 0, stream>>>(A, Wt, bias, res, Cf, Ch, M, N, Np, act); break;
    }
  };

  // ---- weight conversion (every call: deterministic) ----
  wconv(enc_w, encT, 161, 256, 192, 256);
  for (int l = 0; l < 4; ++l) {
    int bse = 10 + 16 * l;
    wconv((const float*)d_in[bse + 0],  wqT[l], 256, 256, 256, 256);
    wconv((const float*)d_in[bse + 2],  wkT[l], 256, 256, 256, 256);
    wconv((const float*)d_in[bse + 4],  wvT[l], 256, 256, 256, 256);
    wconv((const float*)d_in[bse + 6],  woT[l], 256, 256, 256, 256);
    wconv((const float*)d_in[bse + 12], w1T[l], 256, 512, 256, 512);
    wconv((const float*)d_in[bse + 14], w2T[l], 512, 256, 512, 256);
  }
  wconv(hw1, h1T, 256, 128, 256, 128);
  wconv(hw2, h2T, 128, 52, 128, 64);

  // ---- encoder ----
  build_x0_kernel<<<(TOKS * 192 + TB - 1) / TB, TB, 0, stream>>>(af, pt, pi, si, ph_emb, sp_emb, X0);
  gemm(X0, encT, enc_b, nullptr, pre, nullptr, TOKS, 256, 192, 256, 0);
  enc_post_kernel<<<(TOKS * 32 + TB - 1) / TB, TB, 0, stream>>>(pre, encln_s, encln_b, x);

  // ---- transformer layers ----
  for (int l = 0; l < 4; ++l) {
    int bse = 10 + 16 * l;
    const float* bq   = (const float*)d_in[bse + 1];
    const float* bk   = (const float*)d_in[bse + 3];
    const float* bv   = (const float*)d_in[bse + 5];
    const float* bo   = (const float*)d_in[bse + 7];
    const float* ln1s = (const float*)d_in[bse + 8];
    const float* ln1b = (const float*)d_in[bse + 9];
    const float* ln2s = (const float*)d_in[bse + 10];
    const float* ln2b = (const float*)d_in[bse + 11];
    const float* b1   = (const float*)d_in[bse + 13];
    const float* b2   = (const float*)d_in[bse + 15];

    ln_f16_kernel<<<(TOKS * 32 + TB - 1) / TB, TB, 0, stream>>>(x, ln1s, ln1b, H);
    gemm(H, wqT[l], bq, nullptr, nullptr, Qb, TOKS, 256, 256, 256, 0);
    gemm(H, wkT[l], bk, nullptr, nullptr, Kb, TOKS, 256, 256, 256, 0);
    gemm(H, wvT[l], bv, nullptr, nullptr, Vb, TOKS, 256, 256, 256, 0);
    banded_attn_kernel<<<(TOKS * NHEAD + TB - 1) / TB, TB, 0, stream>>>(Qb, Kb, Vb, Ob);
    gemm(Ob, woT[l], bo, x, x, nullptr, TOKS, 256, 256, 256, 0);  // x += O@Wo + bo
    ln_f16_kernel<<<(TOKS * 32 + TB - 1) / TB, TB, 0, stream>>>(x, ln2s, ln2b, H);
    gemm(H, w1T[l], b1, nullptr, nullptr, Fb, TOKS, 512, 256, 512, 1);   // relu
    gemm(Fb, w2T[l], b2, x, x, nullptr, TOKS, 256, 512, 256, 0);         // x += F@W2 + b2
  }

  // ---- head ----
  f32_to_f16_kernel<<<(TOKS * DMODEL + TB - 1) / TB, TB, 0, stream>>>(x, H, TOKS * DMODEL);
  gemm(H, h1T, hb1, nullptr, nullptr, Y1, TOKS, 128, 256, 128, 2);       // gelu
  gemm(Y1, h2T, hb2, nullptr, (float*)d_out, nullptr, TOKS, 52, 128, 64, 3); // sigmoid
}
